// Luban7_29609504539316
// MI455X (gfx1250) — matrix operand order repaired and checked by reading, not compiled, not run on hardware
//
#include <hip/hip_runtime.h>
#include <hip/hip_bf16.h>
#include <math.h>

typedef __attribute__((ext_vector_type(16))) _Float16 v16h;
typedef __attribute__((ext_vector_type(8)))  float    v8f;

#define B_   32
#define T_   120
#define E_   256
#define H_   512
#define G4_  2048          // 4*H
#define C_   20
#define S_   1155
#define M_BT 3840          // B*T
#define MS_  36960         // B*S

// ---------------------------------------------------------------------------
// Fragment-packed layouts (wave32, V_WMMA_F32_16X16X32_F16):
//  A tile (16x32 MxK): lane lo = row M, element e <-> k = ((e>>3)<<4)|(hi<<3)|
//                      (((e>>1)&3)<<1)|(e&1),  hi = lane>>4.
//  B tile (32x16 KxN): lane lo = col N, element e <-> k = ((e>>3)<<4)|(hi<<3)|(e&7).
// Packed storage: tile = 32 lanes x 16 halves contiguous (1024 B). A fragment
// load is then one aligned 32-byte v16h load per lane (2 x b128).
// ---------------------------------------------------------------------------
__device__ __forceinline__ v16h ld_frag(const _Float16* tileBase) {
  return *(const v16h*)(tileBase + ((threadIdx.x & 31) << 4));
}
__device__ __forceinline__ v8f wmma16(v16h a, v16h b, v8f c) {
  return __builtin_amdgcn_wmma_f32_16x16x32_f16(false, a, false, b, (short)0, c,
                                                false, false);
}
// inverse A-map: k_local -> element index (hi = (kl>>3)&1 handled by caller)
__device__ __forceinline__ int a_e(int kl) {
  return ((kl >> 4) << 3) | (((kl >> 1) & 3) << 1) | (kl & 1);
}

// ---------------- prep: embedding gather -> A-frag-packed xh ----------------
// xhp tiles: [tm(240)][kt(8)][lane][e]; m-row = t*32 + b  (time-major)
__global__ void k_embed(const int* __restrict__ tok, const float* __restrict__ emb,
                        _Float16* __restrict__ xhp) {
  int id = blockIdx.x * blockDim.x + threadIdx.x;
  if (id >= M_BT * E_) return;
  int e = id & 15, lane = (id >> 4) & 31, tile = id >> 9;
  int tm = tile >> 3, kt = tile & 7;
  int m = (tm << 4) + (lane & 15);
  int k = (kt << 5) | ((e >> 3) << 4) | (((lane >> 4) & 1) << 3) |
          (((e >> 1) & 3) << 1) | (e & 1);
  int t = m >> 5, b = m & 31;
  xhp[id] = (_Float16)emb[(size_t)tok[b * T_ + t] * E_ + k];
}

// ---------------- prep: weight W (NxK) -> B-frag-packed W^T (KxN) -----------
__global__ void k_packBT(const float* __restrict__ src, _Float16* __restrict__ dst,
                         int K, int N) {
  int id = blockIdx.x * blockDim.x + threadIdx.x;
  if (id >= K * N) return;
  int e = id & 15, lane = (id >> 4) & 31, tile = id >> 9;
  int nt16 = N >> 4;
  int kt = tile / nt16, nt = tile % nt16;
  int k = (kt << 5) + ((e >> 3) << 4) + ((lane >> 4) << 3) + (e & 7);
  int n = (nt << 4) + (lane & 15);
  dst[id] = (_Float16)src[(size_t)n * K + k];
}

// label_w (512x20) -> B-frag-packed, N padded to 32: tiles [kt(16)][nt(2)]
__global__ void k_packLabel(const float* __restrict__ src, _Float16* __restrict__ dst) {
  int id = blockIdx.x * blockDim.x + threadIdx.x;
  if (id >= H_ * 32) return;
  int e = id & 15, lane = (id >> 4) & 31, tile = id >> 9;
  int kt = tile >> 1, nt = tile & 1;
  int k = (kt << 5) + ((e >> 3) << 4) + ((lane >> 4) << 3) + (e & 7);
  int n = (nt << 4) + (lane & 15);
  dst[id] = (_Float16)((n < C_) ? src[(size_t)k * C_ + n] : 0.f);
}

// ---------------- input GEMM: xg = xh @ w_ihT + (b_ih + b_hh) ---------------
// xg stored C-frag-packed: [tM(240)][tN(128)][lane][8 f32]
__global__ void __launch_bounds__(256)
k_ingemm(const _Float16* __restrict__ xh,
         const _Float16* __restrict__ wt0, const _Float16* __restrict__ wt1,
         const float* __restrict__ bi0, const float* __restrict__ bh0,
         const float* __restrict__ bi1, const float* __restrict__ bh1,
         float* __restrict__ xgo0, float* __restrict__ xgo1) {
  int w = blockIdx.x * 8 + (threadIdx.x >> 5);
  int dir = (w >= 240 * 128) ? 1 : 0;
  if (dir) w -= 240 * 128;
  const _Float16* wt = dir ? wt1 : wt0;
  const float* bi = dir ? bi1 : bi0;
  const float* bh = dir ? bh1 : bh0;
  float* outp = dir ? xgo1 : xgo0;
  int tn = w % 128, tm = w / 128;
  v8f acc = {};
#pragma unroll
  for (int kt = 0; kt < 8; ++kt) {
    v16h a = ld_frag(&xh[((size_t)tm * 8 + kt) << 9]);
    v16h b = ld_frag(&wt[((size_t)kt * 128 + tn) << 9]);
    acc = wmma16(a, b, acc);
  }
  int lane = threadIdx.x & 31;
  int n = tn * 16 + (lane & 15);
  float bsum = bi[n] + bh[n];
  float* o = &outp[(((size_t)tm * 128 + tn) * 32 + lane) * 8];
#pragma unroll
  for (int r = 0; r < 8; ++r) o[r] = acc[r] + bsum;
}

// ---------------- LSTM recurrence: 1 workgroup (16 waves) per direction -----
// hbuf: A-frag-packed h state [kt(16)][mt(2)][lane][e]  (32 KB LDS).
// Wave wv owns hidden cols [wv*32, wv*32+32): its 16 acc tiles hold all four
// gates for those (b,j), so cell state never leaves registers.
__global__ void __launch_bounds__(512)
k_lstm(const float* __restrict__ xg0, const float* __restrict__ xg1,
       const _Float16* __restrict__ whh0, const _Float16* __restrict__ whh1,
       _Float16* __restrict__ rnn) {
  __shared__ __align__(32) _Float16 hbuf[B_ * H_];
  const int dir = blockIdx.x;
  const float* xg = dir ? xg1 : xg0;
  const _Float16* whh = dir ? whh1 : whh0;
  const int tid = threadIdx.x;
  const int wv = tid >> 5, lane = tid & 31, lo = lane & 15, hi = lane >> 4;

  for (int i = tid; i < B_ * H_; i += 512) hbuf[i] = (_Float16)0.f;
  float cst[4][8];
#pragma unroll
  for (int a = 0; a < 4; ++a)
#pragma unroll
    for (int r = 0; r < 8; ++r) cst[a][r] = 0.f;
  __syncthreads();

  for (int step = 0; step < T_; ++step) {
    const int tt = dir ? (T_ - 1 - step) : step;
    v8f acc[16];
#pragma unroll
    for (int i = 0; i < 16; ++i) { v8f z = {}; acc[i] = z; }
    for (int kt = 0; kt < 16; ++kt) {
      v16h a0 = ld_frag(&hbuf[(size_t)(kt * 2 + 0) << 9]);
      v16h a1 = ld_frag(&hbuf[(size_t)(kt * 2 + 1) << 9]);
#pragma unroll
      for (int q = 0; q < 4; ++q) {
#pragma unroll
        for (int u = 0; u < 2; ++u) {
          int tN = q * 32 + wv * 2 + u;
          v16h bb = ld_frag(&whh[((size_t)kt * 128 + tN) << 9]);
          int t4 = q * 4 + u * 2;
          acc[t4 + 0] = wmma16(a0, bb, acc[t4 + 0]);
          acc[t4 + 1] = wmma16(a1, bb, acc[t4 + 1]);
        }
      }
      if (kt + 1 < 16)
        __builtin_prefetch(&whh[((size_t)(kt + 1) * 128 + wv * 2) << 9], 0, 1);
    }
    __syncthreads();   // all waves done reading hbuf
#pragma unroll
    for (int mt = 0; mt < 2; ++mt) {
#pragma unroll
      for (int u = 0; u < 2; ++u) {
        int j = wv * 32 + u * 16 + lo;
        // xg C-frag-packed gate pointers (8 contiguous f32 per lane each)
        size_t mbase = ((size_t)(tt * 2 + mt) * 128 + wv * 2 + u) * 32 + lane;
        const float* g0 = &xg[(mbase + (size_t)(0 * 32) * 32) * 8];
        const float* g1 = &xg[(mbase + (size_t)(1 * 32) * 32) * 8];
        const float* g2 = &xg[(mbase + (size_t)(2 * 32) * 32) * 8];
        const float* g3 = &xg[(mbase + (size_t)(3 * 32) * 32) * 8];
#pragma unroll
        for (int r = 0; r < 8; ++r) {
          int b = mt * 16 + hi * 8 + r;
          float gi = acc[0 * 4 + u * 2 + mt][r] + g0[r];
          float gf = acc[1 * 4 + u * 2 + mt][r] + g1[r];
          float gg = acc[2 * 4 + u * 2 + mt][r] + g2[r];
          float go = acc[3 * 4 + u * 2 + mt][r] + g3[r];
          float si = 1.f / (1.f + __expf(-gi));
          float sf = 1.f / (1.f + __expf(-gf));
          float so = 1.f / (1.f + __expf(-go));
          float c = sf * cst[u * 2 + mt][r] + si * tanhf(gg);
          cst[u * 2 + mt][r] = c;
          float h = so * tanhf(c);
          // hbuf (A-frag packed): row=b, k=j
          {
            int kt2 = j >> 5, kl = j & 31;
            int lane2 = (b & 15) | (((kl >> 3) & 1) << 4);
            hbuf[(((kt2 * 2 + (b >> 4)) << 5) + lane2) * 16 + a_e(kl)] = (_Float16)h;
          }
          // rnn (A-frag packed for lin1): row = b*T+tt, k = dir*512 + j
          {
            int m = b * T_ + tt;
            int kr = dir * H_ + j;
            int ktr = kr >> 5, klr = kr & 31;
            int lane3 = (m & 15) | (((klr >> 3) & 1) << 4);
            rnn[(((size_t)(m >> 4) * 32 + ktr) * 32 + lane3) * 16 + a_e(klr)] =
                (_Float16)h;
          }
        }
      }
    }
    __syncthreads();   // hbuf updated before next step
  }
}

// ---------------- lin1 + ReLU: (3840x1024)@(1024x512) -> A-frag-packed h1 ---
__global__ void __launch_bounds__(256)
k_lin1(const _Float16* __restrict__ rnn, const _Float16* __restrict__ w1t,
       const float* __restrict__ b1, _Float16* __restrict__ h1) {
  int w = blockIdx.x * 8 + (threadIdx.x >> 5);
  int tn = w % 32, tm = w / 32;
  v8f acc = {};
#pragma unroll 4
  for (int kt = 0; kt < 32; ++kt) {
    v16h a = ld_frag(&rnn[((size_t)tm * 32 + kt) << 9]);
    v16h b = ld_frag(&w1t[((size_t)kt * 32 + tn) << 9]);
    acc = wmma16(a, b, acc);
  }
  int lane = threadIdx.x & 31, lo = lane & 15, hi = lane >> 4;
  int n = tn * 16 + lo;
  float bb = b1[n];
  int kt2 = n >> 5, kl = n & 31;
  int lane2hi = ((kl >> 3) & 1) << 4;
  int e2 = a_e(kl);
#pragma unroll
  for (int r = 0; r < 8; ++r) {
    float v = acc[r] + bb;
    v = v > 0.f ? v : 0.f;
    int m = tm * 16 + hi * 8 + r;
    h1[(((size_t)tm * 16 + kt2) * 32 + ((m & 15) | lane2hi)) * 16 + e2] = (_Float16)v;
  }
}

// ---------------- lin2: (3840x512)@(512x512) -> row-major fp32 h2 -----------
__global__ void __launch_bounds__(256)
k_lin2(const _Float16* __restrict__ h1, const _Float16* __restrict__ w2t,
       const float* __restrict__ b2, float* __restrict__ h2) {
  int w = blockIdx.x * 8 + (threadIdx.x >> 5);
  int tn = w % 32, tm = w / 32;
  v8f acc = {};
#pragma unroll 4
  for (int kt = 0; kt < 16; ++kt) {
    v16h a = ld_frag(&h1[((size_t)tm * 16 + kt) << 9]);
    v16h b = ld_frag(&w2t[((size_t)kt * 32 + tn) << 9]);
    acc = wmma16(a, b, acc);
  }
  int lane = threadIdx.x & 31, lo = lane & 15, hi = lane >> 4;
  int n = tn * 16 + lo;
  float bb = b2[n];
#pragma unroll
  for (int r = 0; r < 8; ++r)
    h2[(size_t)(tm * 16 + hi * 8 + r) * H_ + n] = acc[r] + bb;
}

// ---------------- LayerNorm (in place, one block per row) -------------------
__global__ void __launch_bounds__(256)
k_ln(float* __restrict__ h2, const float* __restrict__ g, const float* __restrict__ b) {
  __shared__ float red[256], red2[256];
  int row = blockIdx.x, tid = threadIdx.x;
  float s = 0.f, s2 = 0.f;
  for (int j = tid; j < H_; j += 256) {
    float v = h2[(size_t)row * H_ + j];
    s += v; s2 += v * v;
  }
  red[tid] = s; red2[tid] = s2;
  __syncthreads();
  for (int off = 128; off; off >>= 1) {
    if (tid < off) { red[tid] += red[tid + off]; red2[tid] += red2[tid + off]; }
    __syncthreads();
  }
  float mu = red[0] / H_;
  float var = red2[0] / H_ - mu * mu;
  float inv = rsqrtf(var + 1e-5f);
  for (int j = tid; j < H_; j += 256) {
    float v = h2[(size_t)row * H_ + j];
    h2[(size_t)row * H_ + j] = (v - mu) * inv * g[j] + b[j];
  }
}

// ---------------- cumsum over T per (b, feature) ----------------------------
__global__ void k_cumsum(const float* __restrict__ tr, float* __restrict__ cs) {
  int id = blockIdx.x * blockDim.x + threadIdx.x;
  if (id >= B_ * H_) return;
  int b = id >> 9, j = id & 511;
  float acc = 0.f;
  cs[((size_t)b * (T_ + 1) + 0) * H_ + j] = 0.f;
  for (int t = 0; t < T_; ++t) {
    acc += tr[(size_t)(b * T_ + t) * H_ + j];
    cs[((size_t)b * (T_ + 1) + t + 1) * H_ + j] = acc;
  }
}

// ---------------- span pooling + label GEMM (M=36960,K=512,N=20->32) --------
__global__ void __launch_bounds__(128)
k_span(const float* __restrict__ cs, const _Float16* __restrict__ lw,
       const float* __restrict__ lb, float* __restrict__ outp) {
  __shared__ __align__(32) _Float16 stage[32 * H_];   // A-frag packed [kt16][mt2]
  int tid = threadIdx.x;
  int r0 = blockIdx.x * 32;
  for (int idx = tid; idx < 32 * H_; idx += 128) {
    int e = idx & 15, lane = (idx >> 4) & 31, tile = idx >> 9;
    int kt = tile >> 1, mt = tile & 1;
    int i = (mt << 4) + (lane & 15);
    int k = (kt << 5) | ((e >> 3) << 4) | (((lane >> 4) & 1) << 3) |
            (((e >> 1) & 3) << 1) | (e & 1);
    int r = r0 + i;
    int b = r / S_, s = r % S_;
    int beg, len;
    if (s < 1110) { beg = s / 10; len = s % 10 + 1; }
    else {
      int rem = s - 1110, n = 9; beg = 111;
      while (rem >= n) { rem -= n; --n; ++beg; }
      len = rem + 1;
    }
    int en = beg + len;
    float v = (cs[((size_t)b * (T_ + 1) + en) * H_ + k] -
               cs[((size_t)b * (T_ + 1) + beg) * H_ + k]) * (1.f / (float)len);
    stage[idx] = (_Float16)v;
  }
  __syncthreads();
  int wv = tid >> 5, lane = tid & 31, lo = lane & 15, hi = lane >> 4;
  int mt = wv >> 1, u = wv & 1;
  v8f acc = {};
#pragma unroll 4
  for (int kt = 0; kt < 16; ++kt) {
    v16h a = ld_frag(&stage[(size_t)(kt * 2 + mt) << 9]);
    v16h b = ld_frag(&lw[((size_t)kt * 2 + u) << 9]);
    acc = wmma16(a, b, acc);
  }
  int n = u * 16 + lo;
  if (n < C_) {
    float bb = lb[n];
#pragma unroll
    for (int r = 0; r < 8; ++r)
      outp[(size_t)(r0 + mt * 16 + hi * 8 + r) * C_ + n] = acc[r] + bb;
  }
}

// ---------------- log_softmax over axis 0 (per column) ----------------------
__global__ void __launch_bounds__(256)
k_smax_stats(const float* __restrict__ outp, float* __restrict__ stats) {
  __shared__ float red[256 * C_];
  __shared__ float mx[C_];
  int tid = threadIdx.x;
  float lm[C_];
#pragma unroll
  for (int c = 0; c < C_; ++c) lm[c] = -3.4e38f;
  for (int r = tid; r < MS_; r += 256)
#pragma unroll
    for (int c = 0; c < C_; ++c) lm[c] = fmaxf(lm[c], outp[(size_t)r * C_ + c]);
#pragma unroll
  for (int c = 0; c < C_; ++c) red[tid * C_ + c] = lm[c];
  __syncthreads();
  for (int off = 128; off; off >>= 1) {
    if (tid < off)
#pragma unroll
      for (int c = 0; c < C_; ++c)
        red[tid * C_ + c] = fmaxf(red[tid * C_ + c], red[(tid + off) * C_ + c]);
    __syncthreads();
  }
  if (tid < C_) mx[tid] = red[tid];
  __syncthreads();
  float ls[C_];
#pragma unroll
  for (int c = 0; c < C_; ++c) ls[c] = 0.f;
  for (int r = tid; r < MS_; r += 256)
#pragma unroll
    for (int c = 0; c < C_; ++c)
      ls[c] += __expf(outp[(size_t)r * C_ + c] - mx[c]);
#pragma unroll
  for (int c = 0; c < C_; ++c) red[tid * C_ + c] = ls[c];
  __syncthreads();
  for (int off = 128; off; off >>= 1) {
    if (tid < off)
#pragma unroll
      for (int c = 0; c < C_; ++c) red[tid * C_ + c] += red[(tid + off) * C_ + c];
    __syncthreads();
  }
  if (tid < C_) stats[tid] = mx[tid] + __logf(red[tid]);
}

__global__ void k_smax_apply(float* __restrict__ outp, const float* __restrict__ stats) {
  size_t id = (size_t)blockIdx.x * blockDim.x + threadIdx.x;
  if (id >= (size_t)MS_ * C_) return;
  int c = (int)(id % C_);
  outp[id] -= stats[c];
}

// ---------------- launch ----------------------------------------------------
extern "C" void kernel_launch(void* const* d_in, const int* in_sizes, int n_in,
                              void* d_out, int out_size, void* d_ws, size_t ws_size,
                              hipStream_t stream) {
  (void)in_sizes; (void)n_in; (void)out_size; (void)ws_size;
  const int*   tokens  = (const int*)d_in[0];
  const float* emb     = (const float*)d_in[1];
  const float* w_ih_f  = (const float*)d_in[2];
  const float* w_hh_f  = (const float*)d_in[3];
  const float* b_ih_f  = (const float*)d_in[4];
  const float* b_hh_f  = (const float*)d_in[5];
  const float* w_ih_b  = (const float*)d_in[6];
  const float* w_hh_b  = (const float*)d_in[7];
  const float* b_ih_b  = (const float*)d_in[8];
  const float* b_hh_b  = (const float*)d_in[9];
  const float* lin1_w  = (const float*)d_in[10];
  const float* lin1_b  = (const float*)d_in[11];
  const float* lin2_w  = (const float*)d_in[12];
  const float* lin2_b  = (const float*)d_in[13];
  const float* ln_g    = (const float*)d_in[14];
  const float* ln_b    = (const float*)d_in[15];
  const float* label_w = (const float*)d_in[16];
  const float* label_b = (const float*)d_in[17];
  float* outp = (float*)d_out;
  char* ws = (char*)d_ws;

  // workspace layout (bytes, 256-aligned); XG0/XG1 reused as H2/CS later
  _Float16* XHP   = (_Float16*)(ws + 0);          // 240x8 A-tiles
  _Float16* WIHP0 = (_Float16*)(ws + 1966080);    // 8x128 B-tiles
  _Float16* WIHP1 = (_Float16*)(ws + 3014656);
  _Float16* WHHP0 = (_Float16*)(ws + 4063232);    // 16x128 B-tiles
  _Float16* WHHP1 = (_Float16*)(ws + 6160384);
  _Float16* LIN1P = (_Float16*)(ws + 8257536);    // 32x32 B-tiles
  _Float16* LIN2P = (_Float16*)(ws + 9306112);    // 16x32 B-tiles
  _Float16* LABP  = (_Float16*)(ws + 9830400);    // 16x2 B-tiles (padded)
  float*    STATS = (float*)   (ws + 9863168);    // 20 f32
  float*    XG0   = (float*)   (ws + 9863424);    // 240x128 C-tiles f32
  float*    XG1   = (float*)   (ws + 41320704);
  _Float16* RNNP  = (_Float16*)(ws + 72777984);   // 240x32 A-tiles
  _Float16* H1P   = (_Float16*)(ws + 80642304);   // 240x16 A-tiles
  float*    H2    = XG0;                          // 3840x512 f32 (reuse)
  float*    CS    = XG1;                          // 32x121x512 f32 (reuse)

  k_embed<<<(M_BT * E_ + 255) / 256, 256, 0, stream>>>(tokens, emb, XHP);
  k_packBT<<<(G4_ * E_ + 255) / 256, 256, 0, stream>>>(w_ih_f, WIHP0, E_, G4_);
  k_packBT<<<(G4_ * E_ + 255) / 256, 256, 0, stream>>>(w_ih_b, WIHP1, E_, G4_);
  k_packBT<<<(G4_ * H_ + 255) / 256, 256, 0, stream>>>(w_hh_f, WHHP0, H_, G4_);
  k_packBT<<<(G4_ * H_ + 255) / 256, 256, 0, stream>>>(w_hh_b, WHHP1, H_, G4_);
  k_packBT<<<(H_ * 2 * H_ + 255) / 256, 256, 0, stream>>>(lin1_w, LIN1P, 2 * H_, H_);
  k_packBT<<<(H_ * H_ + 255) / 256, 256, 0, stream>>>(lin2_w, LIN2P, H_, H_);
  k_packLabel<<<(H_ * 32 + 255) / 256, 256, 0, stream>>>(label_w, LABP);

  k_ingemm<<<7680, 256, 0, stream>>>(XHP, WIHP0, WIHP1, b_ih_f, b_hh_f,
                                     b_ih_b, b_hh_b, XG0, XG1);
  k_lstm<<<2, 512, 0, stream>>>(XG0, XG1, WHHP0, WHHP1, RNNP);

  k_lin1<<<960, 256, 0, stream>>>(RNNP, LIN1P, lin1_b, H1P);
  k_lin2<<<960, 256, 0, stream>>>(H1P, LIN2P, lin2_b, H2);
  k_ln<<<M_BT, 256, 0, stream>>>(H2, ln_g, ln_b);
  k_cumsum<<<(B_ * H_ + 255) / 256, 256, 0, stream>>>(H2, CS);
  k_span<<<MS_ / 32, 128, 0, stream>>>(CS, LABP, label_b, outp);
  k_smax_stats<<<1, 256, 0, stream>>>(outp, STATS);
  k_smax_apply<<<((size_t)MS_ * C_ + 255) / 256, 256, 0, stream>>>(outp, STATS);
}
